// MemoryTokenLayer_74552042324781
// MI455X (gfx1250) — matrix operand order repaired and checked by reading, not compiled, not run on hardware
//
#include <hip/hip_runtime.h>

#define D_MODEL 1024
#define N_HEADS 16
#define HEAD_DIM 64
#define N_MEM 16
#define T_SEQ 2048
#define B_SZ 2
#define S_SEQ 2064   // N_MEM + T_SEQ
#define S_PAD 2080   // padded to multiple of 32 for the attention K loop
#define BH_TOT (B_SZ * N_HEADS)

typedef __attribute__((ext_vector_type(16))) __bf16 bf16x16;
typedef __attribute__((ext_vector_type(8)))  float  f32x8;

union Frag16 { unsigned int u[8]; bf16x16 v; };

__device__ __forceinline__ f32x8 zero8() {
  f32x8 z;
#pragma unroll
  for (int i = 0; i < 8; ++i) z[i] = 0.0f;
  return z;
}

__device__ __forceinline__ unsigned short f2bf(float f) {
  unsigned int u = __builtin_bit_cast(unsigned int, f);
  u += 0x7FFFu + ((u >> 16) & 1u);   // round-to-nearest-even
  return (unsigned short)(u >> 16);
}

__device__ __forceinline__ float bf2f(unsigned short h) {
  unsigned int u = ((unsigned int)h) << 16;
  return __builtin_bit_cast(float, u);
}

// ---- WMMA fragment loaders (layouts per CDNA5 ISA 7.12.2, wave32) ----

// A fragment: 16x32 bf16 tile; rows row0..row0+15 (row-major, stride lda elems),
// K columns k0..k0+31.  Lane L: M = L&15, half = L>>4.
// VGPR i holds K = (i/4)*16 + 8*half + 2*(i%4) .. +1  (dword loads).
__device__ __forceinline__ bf16x16 load_a_frag(const unsigned short* A, int lda,
                                               int row0, int k0, int lane) {
  int row = row0 + (lane & 15);
  int half = lane >> 4;
  const unsigned short* p = A + (size_t)row * lda + k0 + (half << 3);
  Frag16 f;
#pragma unroll
  for (int i = 0; i < 8; ++i) {
    int koff = ((i >> 2) << 4) + ((i & 3) << 1);
    f.u[i] = *reinterpret_cast<const unsigned int*>(p + koff);
  }
  return f.v;
}

// B fragment: 32x16 where B[k][n] = W[n0+n][k0+k] with W row-major (stride ldw).
// Lane L: N = L&15, half = L>>4.  VGPR j holds K = (j/4)*16 + 8*half + 2*(j%4) .. +1.
__device__ __forceinline__ bf16x16 load_b_frag(const unsigned short* W, int ldw,
                                               int n0, int k0, int lane) {
  int n = n0 + (lane & 15);
  int half = lane >> 4;
  const unsigned short* p = W + (size_t)n * ldw + k0 + (half << 3);
  Frag16 f;
#pragma unroll
  for (int j = 0; j < 8; ++j)
    f.u[j] = *reinterpret_cast<const unsigned int*>(p + ((j >> 2) << 4) + ((j & 3) << 1));
  return f.v;
}

// A fragment from a 16x32 bf16 tile in LDS (row-major, 32-element rows).
__device__ __forceinline__ bf16x16 load_p_frag(const unsigned short* pb, int lane) {
  int row = lane & 15;
  int half = lane >> 4;
  const unsigned short* p = pb + row * 32 + (half << 3);
  Frag16 f;
#pragma unroll
  for (int i = 0; i < 8; ++i) {
    int koff = ((i >> 2) << 4) + ((i & 3) << 1);
    f.u[i] = *reinterpret_cast<const unsigned int*>(p + koff);
  }
  return f.v;
}

__device__ __forceinline__ f32x8 wmma_bf16(bf16x16 a, bf16x16 b, f32x8 c) {
  return __builtin_amdgcn_wmma_f32_16x16x32_bf16(false, a, false, b,
                                                 (short)0, c, false, false);
}

// ---------------- fp32 -> bf16 weight conversion ----------------
__global__ __launch_bounds__(256) void f32_to_bf16_kernel(
    const float* __restrict__ in, unsigned short* __restrict__ out, int n) {
  int i = blockIdx.x * 256 + threadIdx.x;
  if (i < n) out[i] = f2bf(in[i]);
}

// ---------------- LayerNorm of [mem ; x] -> xm (bf16, B*S x D) ----------------
__global__ __launch_bounds__(128) void ln_concat_kernel(
    const float* __restrict__ x, const float* __restrict__ mem,
    const float* __restrict__ ng, const float* __restrict__ nb,
    const float* __restrict__ mg, const float* __restrict__ mb_,
    unsigned short* __restrict__ xm) {
  int row = blockIdx.x;                 // 0 .. B*S-1
  int b = row / S_SEQ;
  int s = row - b * S_SEQ;
  const float* src; const float* g; const float* be;
  if (s < N_MEM) { src = mem + (size_t)s * D_MODEL; g = mg; be = mb_; }
  else { src = x + ((size_t)b * T_SEQ + (s - N_MEM)) * D_MODEL; g = ng; be = nb; }

  float sum = 0.f, sq = 0.f;
  for (int d = threadIdx.x; d < D_MODEL; d += 128) {
    float v = src[d]; sum += v; sq += v * v;
  }
#pragma unroll
  for (int off = 16; off; off >>= 1) {
    sum += __shfl_xor(sum, off, 32);
    sq  += __shfl_xor(sq, off, 32);
  }
  __shared__ float rs[4], rq[4];
  int wave = threadIdx.x >> 5;
  if ((threadIdx.x & 31) == 0) { rs[wave] = sum; rq[wave] = sq; }
  __syncthreads();
  sum = rs[0] + rs[1] + rs[2] + rs[3];
  sq  = rq[0] + rq[1] + rq[2] + rq[3];
  float mu = sum * (1.0f / D_MODEL);
  float var = sq * (1.0f / D_MODEL) - mu * mu;
  float rstd = rsqrtf(var + 1e-5f);
  unsigned short* dst = xm + (size_t)row * D_MODEL;
  for (int d = threadIdx.x; d < D_MODEL; d += 128)
    dst[d] = f2bf((src[d] - mu) * rstd * g[d] + be[d]);
}

// ---------------- QKV GEMM: [B*S,1024] x [3072,1024]^T + bias ----------------
// 32x64 tile per wave (2 A frags x 4 B frags, 8 accumulators).
// Scatters q,k -> (bh, s, hd) bf16 and v -> (bh, hd, s) transposed bf16.
__global__ __launch_bounds__(128) void qkv_gemm_kernel(
    const unsigned short* __restrict__ xm, const unsigned short* __restrict__ wq,
    const float* __restrict__ qbias,
    unsigned short* __restrict__ qo, unsigned short* __restrict__ ko,
    unsigned short* __restrict__ vo) {
  int lane = threadIdx.x & 31;
  int wave = threadIdx.x >> 5;
  int mt = blockIdx.x;                      // 129 tiles of 32 rows
  int m0 = mt * 32;
  int n0 = blockIdx.y * 256 + wave * 64;    // 12 * 256 = 3072 cols

  f32x8 acc0[4], acc1[4];
#pragma unroll
  for (int t = 0; t < 4; ++t) { acc0[t] = zero8(); acc1[t] = zero8(); }

  for (int k0 = 0; k0 < D_MODEL; k0 += 32) {
    bf16x16 a0 = load_a_frag(xm, D_MODEL, m0, k0, lane);
    bf16x16 a1 = load_a_frag(xm, D_MODEL, m0 + 16, k0, lane);
    if (k0 + 32 < D_MODEL)  // prefetch next K-slab of A rows (global_prefetch_b8)
      __builtin_prefetch(xm + (size_t)(m0 + lane) * D_MODEL + k0 + 32, 0, 1);
#pragma unroll
    for (int t = 0; t < 4; ++t) {
      bf16x16 b = load_b_frag(wq, D_MODEL, n0 + t * 16, k0, lane);
      acc0[t] = wmma_bf16(a0, b, acc0[t]);
      acc1[t] = wmma_bf16(a1, b, acc1[t]);
    }
  }

  int nn = lane & 15, half = lane >> 4;
#pragma unroll
  for (int t = 0; t < 4; ++t) {
    int n = n0 + t * 16 + nn;
    int which = n >> 10;          // 0=q 1=k 2=v
    int h = (n >> 6) & 15;
    int hd = n & 63;
    float bias = qbias[n];
#pragma unroll
    for (int g = 0; g < 2; ++g) {
#pragma unroll
      for (int r = 0; r < 8; ++r) {
        int m = m0 + g * 16 + r + (half << 3);
        int bi = m / S_SEQ;
        int s = m - bi * S_SEQ;
        int bh = bi * N_HEADS + h;
        float v = (g == 0 ? acc0[t][r] : acc1[t][r]) + bias;
        unsigned short val = f2bf(v);
        if (which == 0)      qo[((size_t)bh * S_PAD + s) * HEAD_DIM + hd] = val;
        else if (which == 1) ko[((size_t)bh * S_PAD + s) * HEAD_DIM + hd] = val;
        else                 vo[((size_t)bh * HEAD_DIM + hd) * S_PAD + s] = val;
      }
    }
  }
}

// ---------------- zero the padding rows of K and V^T ----------------
__global__ __launch_bounds__(256) void zero_pads_kernel(
    unsigned short* __restrict__ ko, unsigned short* __restrict__ vo) {
  int i = blockIdx.x * 256 + threadIdx.x;   // 65536 threads
  if (i < 32768) {
    int hd = i & 63; int r = (i >> 6) & 15; int bh = i >> 10;
    ko[((size_t)bh * S_PAD + (S_SEQ + r)) * HEAD_DIM + hd] = 0;
  } else {
    int j = i - 32768;
    int c = j & 15; int hd = (j >> 4) & 63; int bh = j >> 10;
    vo[((size_t)bh * HEAD_DIM + hd) * S_PAD + (S_SEQ + c)] = 0;
  }
}

// ---------------- RoPE applied in-place to q and k ----------------
__global__ __launch_bounds__(256) void rope_kernel(
    unsigned short* __restrict__ q, unsigned short* __restrict__ k) {
  int idx = blockIdx.x * 256 + threadIdx.x;
  const int total = BH_TOT * S_SEQ * (HEAD_DIM / 2);
  if (idx >= total) return;
  int i = idx & 31;                  // freq index 0..31
  int s = (idx >> 5) % S_SEQ;
  int bh = idx / (S_SEQ * 32);
  size_t base = ((size_t)bh * S_PAD + s) * HEAD_DIM;
  // inv_freq = theta^(-2i/64) = exp(-ln(10000) * 2i/64)
  float inv = expf(-9.210340372f * (float)(2 * i) * (1.0f / 64.0f));
  float ang = (float)s * inv;
  float sn, cs;
  sincosf(ang, &sn, &cs);
  float a1 = bf2f(q[base + i]),      a2 = bf2f(q[base + i + 32]);
  q[base + i]      = f2bf(a1 * cs - a2 * sn);
  q[base + i + 32] = f2bf(a2 * cs + a1 * sn);
  float b1 = bf2f(k[base + i]),      b2 = bf2f(k[base + i + 32]);
  k[base + i]      = f2bf(b1 * cs - b2 * sn);
  k[base + i + 32] = f2bf(b2 * cs + b1 * sn);
}

// ---------------- causal flash attention, one wave per (bh, q-tile) ----------------
__global__ __launch_bounds__(32) void attn_kernel(
    const unsigned short* __restrict__ q, const unsigned short* __restrict__ k,
    const unsigned short* __restrict__ vt, unsigned short* __restrict__ ao) {
  int qt = blockIdx.x;               // 129 q-tiles of 16 rows
  int bh = blockIdx.y;               // 32 (b,h) pairs
  int lane = threadIdx.x & 31;
  int nn = lane & 15, half = lane >> 4;

  const unsigned short* qp = q  + (size_t)bh * S_PAD * HEAD_DIM;
  const unsigned short* kp = k  + (size_t)bh * S_PAD * HEAD_DIM;
  const unsigned short* vp = vt + (size_t)bh * HEAD_DIM * S_PAD;

  bf16x16 qa0 = load_a_frag(qp, HEAD_DIM, qt * 16, 0, lane);
  bf16x16 qa1 = load_a_frag(qp, HEAD_DIM, qt * 16, 32, lane);

  f32x8 o0 = zero8(), o1 = zero8(), o2 = zero8(), o3 = zero8();
  float mi[8], li[8];
#pragma unroll
  for (int r = 0; r < 8; ++r) { mi[r] = -1e30f; li[r] = 0.f; }

  __shared__ __align__(16) unsigned short pb[16 * 32];

  int qbase = qt * 16;
  int niter = (qbase + 16 + 31) >> 5;     // 32-wide key tiles, causal bound
  for (int it = 0; it < niter; ++it) {
    int kb = it << 5;
    // ---- scores: two 16x16 tiles over hd=64 (2 WMMAs each) ----
    f32x8 s0 = zero8(), s1 = zero8();
    s0 = wmma_bf16(qa0, load_b_frag(kp, HEAD_DIM, kb, 0, lane), s0);
    s0 = wmma_bf16(qa1, load_b_frag(kp, HEAD_DIM, kb, 32, lane), s0);
    s1 = wmma_bf16(qa0, load_b_frag(kp, HEAD_DIM, kb + 16, 0, lane), s1);
    s1 = wmma_bf16(qa1, load_b_frag(kp, HEAD_DIM, kb + 16, 32, lane), s1);

    // ---- online softmax on the C-fragment layout ----
#pragma unroll
    for (int r = 0; r < 8; ++r) {
      int qpos = qbase + r + (half << 3);
      float a0 = (kb + nn      > qpos) ? -1e30f : s0[r] * 0.125f;
      float a1 = (kb + 16 + nn > qpos) ? -1e30f : s1[r] * 0.125f;
      float mx = fmaxf(a0, a1);
#pragma unroll
      for (int off = 8; off; off >>= 1) mx = fmaxf(mx, __shfl_xor(mx, off, 32));
      mx = fmaxf(mx, mi[r]);
      float corr = __expf(mi[r] - mx);
      float p0 = __expf(a0 - mx);
      float p1 = __expf(a1 - mx);
      float rsum = p0 + p1;
#pragma unroll
      for (int off = 8; off; off >>= 1) rsum += __shfl_xor(rsum, off, 32);
      li[r] = li[r] * corr + rsum;
      mi[r] = mx;
      o0[r] *= corr; o1[r] *= corr; o2[r] *= corr; o3[r] *= corr;
      int row = r + (half << 3);
      pb[row * 32 + nn]      = f2bf(p0);
      pb[row * 32 + 16 + nn] = f2bf(p1);
    }

    // ---- O += P(16x32) @ V(32x64): 4 WMMAs ----
    bf16x16 pa = load_p_frag(pb, lane);
    o0 = wmma_bf16(pa, load_b_frag(vp, S_PAD, 0,  kb, lane), o0);
    o1 = wmma_bf16(pa, load_b_frag(vp, S_PAD, 16, kb, lane), o1);
    o2 = wmma_bf16(pa, load_b_frag(vp, S_PAD, 32, kb, lane), o2);
    o3 = wmma_bf16(pa, load_b_frag(vp, S_PAD, 48, kb, lane), o3);
  }

  // ---- normalize and store to (B,S,D) bf16 ----
  int bi = bh >> 4, h = bh & 15;
#pragma unroll
  for (int r = 0; r < 8; ++r) {
    int s = qbase + r + (half << 3);
    float inv = 1.0f / li[r];
    size_t base = ((size_t)(bi * S_SEQ + s)) * D_MODEL + h * HEAD_DIM;
    ao[base + nn]      = f2bf(o0[r] * inv);
    ao[base + 16 + nn] = f2bf(o1[r] * inv);
    ao[base + 32 + nn] = f2bf(o2[r] * inv);
    ao[base + 48 + nn] = f2bf(o3[r] * inv);
  }
}

// ---------------- output projection + bias + residual ----------------
// 32x64 tile per wave.
__global__ __launch_bounds__(128) void outproj_kernel(
    const unsigned short* __restrict__ ao, const unsigned short* __restrict__ wo,
    const float* __restrict__ ob, const float* __restrict__ x,
    float* __restrict__ out) {
  int lane = threadIdx.x & 31;
  int wave = threadIdx.x >> 5;
  int mt = blockIdx.x;                      // 128 tiles of 32 rows over (b,t)
  int m0 = mt * 32;
  int bi = m0 >> 11;
  int t0 = m0 & 2047;
  int arow0 = bi * S_SEQ + N_MEM + t0;      // skip mem rows in ao
  int n0 = blockIdx.y * 256 + wave * 64;

  f32x8 acc0[4], acc1[4];
#pragma unroll
  for (int t = 0; t < 4; ++t) { acc0[t] = zero8(); acc1[t] = zero8(); }

  for (int k0 = 0; k0 < D_MODEL; k0 += 32) {
    bf16x16 a0 = load_a_frag(ao, D_MODEL, arow0, k0, lane);
    bf16x16 a1 = load_a_frag(ao, D_MODEL, arow0 + 16, k0, lane);
    if (k0 + 32 < D_MODEL)
      __builtin_prefetch(ao + (size_t)(arow0 + lane) * D_MODEL + k0 + 32, 0, 1);
#pragma unroll
    for (int t = 0; t < 4; ++t) {
      bf16x16 b = load_b_frag(wo, D_MODEL, n0 + t * 16, k0, lane);
      acc0[t] = wmma_bf16(a0, b, acc0[t]);
      acc1[t] = wmma_bf16(a1, b, acc1[t]);
    }
  }

  int nn = lane & 15, half = lane >> 4;
#pragma unroll
  for (int t = 0; t < 4; ++t) {
    int n = n0 + t * 16 + nn;
    float bias = ob[n];
#pragma unroll
    for (int g = 0; g < 2; ++g) {
#pragma unroll
      for (int r = 0; r < 8; ++r) {
        int trow = t0 + g * 16 + r + (half << 3);
        size_t off = ((size_t)bi * T_SEQ + trow) * D_MODEL + n;
        out[off] = (g == 0 ? acc0[t][r] : acc1[t][r]) + bias + x[off];
      }
    }
  }
}

extern "C" void kernel_launch(void* const* d_in, const int* in_sizes, int n_in,
                              void* d_out, int out_size, void* d_ws, size_t ws_size,
                              hipStream_t stream) {
  const float* x       = (const float*)d_in[0];
  const float* mem     = (const float*)d_in[1];
  const float* qkv_w   = (const float*)d_in[2];
  const float* qkv_b   = (const float*)d_in[3];
  const float* out_w   = (const float*)d_in[4];
  const float* out_b   = (const float*)d_in[5];
  const float* norm_g  = (const float*)d_in[6];
  const float* norm_b  = (const float*)d_in[7];
  const float* normm_g = (const float*)d_in[8];
  const float* normm_b = (const float*)d_in[9];
  float* out = (float*)d_out;
  (void)in_sizes; (void)n_in; (void)out_size; (void)ws_size;

  char* ws = (char*)d_ws;
  size_t off = 0;
  auto alloc = [&](size_t bytes) -> void* {
    void* p = ws + off;
    off = (off + bytes + 255) & ~(size_t)255;
    return p;
  };

  unsigned short* wq = (unsigned short*)alloc((size_t)3072 * 1024 * 2);
  unsigned short* wo = (unsigned short*)alloc((size_t)1024 * 1024 * 2);
  unsigned short* xm = (unsigned short*)alloc((size_t)(B_SZ * S_SEQ) * D_MODEL * 2);
  unsigned short* qb = (unsigned short*)alloc((size_t)BH_TOT * S_PAD * HEAD_DIM * 2);
  unsigned short* kb = (unsigned short*)alloc((size_t)BH_TOT * S_PAD * HEAD_DIM * 2);
  unsigned short* vt = (unsigned short*)alloc((size_t)BH_TOT * HEAD_DIM * S_PAD * 2);
  unsigned short* ao = (unsigned short*)alloc((size_t)(B_SZ * S_SEQ) * D_MODEL * 2);

  f32_to_bf16_kernel<<<(3145728 + 255) / 256, 256, 0, stream>>>(qkv_w, wq, 3145728);
  f32_to_bf16_kernel<<<(1048576 + 255) / 256, 256, 0, stream>>>(out_w, wo, 1048576);
  ln_concat_kernel<<<B_SZ * S_SEQ, 128, 0, stream>>>(x, mem, norm_g, norm_b,
                                                     normm_g, normm_b, xm);
  qkv_gemm_kernel<<<dim3(129, 12), 128, 0, stream>>>(xm, wq, qkv_b, qb, kb, vt);
  zero_pads_kernel<<<65536 / 256, 256, 0, stream>>>(kb, vt);
  rope_kernel<<<(BH_TOT * S_SEQ * 32 + 255) / 256, 256, 0, stream>>>(qb, kb);
  attn_kernel<<<dim3(S_SEQ / 16, BH_TOT), 32, 0, stream>>>(qb, kb, vt, ao);
  outproj_kernel<<<dim3(128, 4), 128, 0, stream>>>(ao, wo, out_b, x, out);
}
